// Block_30777735643865
// MI455X (gfx1250) — compile-verified, matrix operand order checked by reading, not run on hardware
//
#include <hip/hip_runtime.h>
#include <math.h>

// ---------------- problem constants ----------------
#define BB   2
#define SS   2048
#define EE   768
#define HH   12
#define DD   64
#define FFF  3072
#define MM   (BB * SS)        // 4096 rows

typedef _Float16 v16h __attribute__((ext_vector_type(16)));
typedef _Float16 v8h  __attribute__((ext_vector_type(8)));
typedef float    v8f  __attribute__((ext_vector_type(8)));

__device__ __forceinline__ v8f wmma_f16(v16h a, v16h b, v8f c) {
    // (neg_a, A, neg_b, B, c_mod, C, reuse_a, reuse_b)
    return __builtin_amdgcn_wmma_f32_16x16x32_f16(false, a, false, b, (short)0, c, false, false);
}

__device__ __forceinline__ v16h ld_frag(const _Float16* p0, const _Float16* p1) {
    v8h lo = *(const v8h*)p0;
    v8h hi = *(const v8h*)p1;
    return __builtin_shufflevector(lo, hi, 0,1,2,3,4,5,6,7,8,9,10,11,12,13,14,15);
}

// async 16B global -> LDS copy (ASYNCcnt-tracked DMA, CDNA5)
__device__ __forceinline__ void async_ld_b128(unsigned lds_off, const void* gaddr) {
    asm volatile("global_load_async_to_lds_b128 %0, %1, off"
                 :: "v"(lds_off), "v"(gaddr) : "memory");
}
__device__ __forceinline__ void wait_async0() {
    asm volatile("s_wait_asynccnt 0" ::: "memory");
}

// ---------------- f32 -> f16 convert ----------------
__global__ void cvt_f16_kernel(const float* __restrict__ in, _Float16* __restrict__ out, int n) {
    int i = blockIdx.x * 256 + threadIdx.x;
    if (i < n) out[i] = (_Float16)in[i];
}

// ---------------- row LayerNorm -> f16 ----------------
__global__ void ln_f16_kernel(const float* __restrict__ x,
                              const float* __restrict__ gamma,
                              const float* __restrict__ beta,
                              _Float16* __restrict__ out) {
    __shared__ float s1[256], s2[256];
    const int row = blockIdx.x;
    const float* xr = x + (size_t)row * EE;
    float v[3], sum = 0.f, sq = 0.f;
#pragma unroll
    for (int i = 0; i < 3; ++i) {
        v[i] = xr[threadIdx.x + i * 256];
        sum += v[i];
        sq  += v[i] * v[i];
    }
    s1[threadIdx.x] = sum; s2[threadIdx.x] = sq;
    __syncthreads();
    for (int s = 128; s > 0; s >>= 1) {
        if (threadIdx.x < s) {
            s1[threadIdx.x] += s1[threadIdx.x + s];
            s2[threadIdx.x] += s2[threadIdx.x + s];
        }
        __syncthreads();
    }
    const float mu = s1[0] * (1.0f / EE);
    const float var = s2[0] * (1.0f / EE) - mu * mu;
    const float rs = rsqrtf(var + 1e-5f);
    _Float16* orow = out + (size_t)row * EE;
#pragma unroll
    for (int i = 0; i < 3; ++i) {
        int c = threadIdx.x + i * 256;
        orow[c] = (_Float16)((v[i] - mu) * rs * gamma[c] + beta[c]);
    }
}

// ---------------- WMMA GEMM: C[M,N] = A[M,K] @ W[N,K]^T ----------------
// LDS-staged via async global->LDS DMA, double buffered.
// Block: 256 thr = 8 waves, tile 128M x 64N; wave tile 32x32 (4 WMMAs/step).
// mode 0: outH f16 row-major        (no bias)
// mode 1: outH f16, V-transposed to [B,H,D,S]  (no bias)
// mode 2: outF f32 = resid + bias + C
// mode 3: outH f16 = gelu(bias + C)   (exact erf)
__global__ void gemm_wmma_kernel(const _Float16* __restrict__ A,
                                 const _Float16* __restrict__ W,
                                 const float* __restrict__ bias,
                                 const float* __restrict__ resid,
                                 float* __restrict__ outF,
                                 _Float16* __restrict__ outH,
                                 int M, int N, int K, int mode) {
    __shared__ _Float16 Abuf[2][128 * 32];   // 8 KB each
    __shared__ _Float16 Bbuf[2][64 * 32];    // 4 KB each

    const int lane = threadIdx.x & 31;
    const int wave = threadIdx.x >> 5;          // 0..7 -> 4M x 2N
    const int mW = (wave >> 1) * 32;            // local M offset of wave tile
    const int nW = (wave & 1) * 32;             // local N offset of wave tile
    const int mBlk = blockIdx.y * 128;
    const int nBlk = blockIdx.x * 64;
    const int ko = (lane & 16) ? 8 : 0;         // K sub-offsets {ko, ko+16}
    const int col = lane & 15;
    const int half = lane >> 4;

    // stage one 32-wide K slab: A 128x32 (2 chunks/thread) + B 64x32 (1 chunk/thread)
    auto stage = [&](int kk, int bufIdx) {
#pragma unroll
        for (int i = 0; i < 2; ++i) {
            int c = threadIdx.x + i * 256;      // 0..511
            int row = c >> 2, q = c & 3;
            const _Float16* g = A + (size_t)(mBlk + row) * K + kk + q * 8;
            async_ld_b128((unsigned)(unsigned long long)&Abuf[bufIdx][row * 32 + q * 8], g);
        }
        {
            int c = threadIdx.x;                // 0..255
            int row = c >> 2, q = c & 3;
            const _Float16* g = W + (size_t)(nBlk + row) * K + kk + q * 8;
            async_ld_b128((unsigned)(unsigned long long)&Bbuf[bufIdx][row * 32 + q * 8], g);
        }
    };

    v8f c00 = {}, c01 = {}, c10 = {}, c11 = {};

    const int nsteps = K >> 5;
    stage(0, 0);
    wait_async0();
    __syncthreads();

    for (int s = 0; s < nsteps; ++s) {
        const int cur = s & 1;
        if (s + 1 < nsteps) stage((s + 1) << 5, cur ^ 1);

        const _Float16* ab = &Abuf[cur][0];
        const _Float16* bb = &Bbuf[cur][0];
        v16h a0 = ld_frag(ab + (mW + col) * 32 + ko,      ab + (mW + col) * 32 + ko + 16);
        v16h a1 = ld_frag(ab + (mW + 16 + col) * 32 + ko, ab + (mW + 16 + col) * 32 + ko + 16);
        v16h b0 = ld_frag(bb + (nW + col) * 32 + ko,      bb + (nW + col) * 32 + ko + 16);
        v16h b1 = ld_frag(bb + (nW + 16 + col) * 32 + ko, bb + (nW + 16 + col) * 32 + ko + 16);
        c00 = wmma_f16(a0, b0, c00);
        c01 = wmma_f16(a0, b1, c01);
        c10 = wmma_f16(a1, b0, c10);
        c11 = wmma_f16(a1, b1, c11);

        wait_async0();        // our next-slab DMAs have landed
        __syncthreads();      // everyone done reading cur before it is overwritten
    }

    // ---------------- writeback ----------------
#pragma unroll
    for (int i = 0; i < 2; ++i) {
#pragma unroll
        for (int r = 0; r < 8; ++r) {
            int m = mBlk + mW + i * 16 + half * 8 + r;
#pragma unroll
            for (int j = 0; j < 2; ++j) {
                int n = nBlk + nW + j * 16 + col;
                float v;
                if (i == 0) v = (j == 0) ? c00[r] : c01[r];
                else        v = (j == 0) ? c10[r] : c11[r];
                if (mode == 0) {
                    outH[(size_t)m * N + n] = (_Float16)v;
                } else if (mode == 1) {
                    int b = m >> 11, sIdx = m & (SS - 1);
                    int h = n >> 6,  d = n & (DD - 1);
                    outH[(((size_t)b * HH + h) * DD + d) * SS + sIdx] = (_Float16)v;
                } else if (mode == 2) {
                    v += bias[n];
                    outF[(size_t)m * N + n] = resid[(size_t)m * N + n] + v;
                } else {
                    v += bias[n];
                    v = 0.5f * v * (1.0f + erff(v * 0.70710678118654752f));
                    outH[(size_t)m * N + n] = (_Float16)v;
                }
            }
        }
    }
}

// ---------------- flash attention (causal) ----------------
// q16,k16: [B,S,H,D] f16 row-major ( == [M,E] );  vt16: [B,H,D,S] f16
// attn16 out: [B,S,H,D] f16 ( == [M,E] )
__global__ void flash_attn_kernel(const _Float16* __restrict__ q16,
                                  const _Float16* __restrict__ k16,
                                  const _Float16* __restrict__ vt16,
                                  _Float16* __restrict__ attn16) {
    __shared__ _Float16 pl[8 * 16 * 32];   // per-wave 16x32 P tile

    const int lane = threadIdx.x & 31;
    const int wave = threadIdx.x >> 5;
    const int half = lane >> 4;
    const int col  = lane & 15;
    const int ko   = (lane & 16) ? 8 : 0;

    const int bh = blockIdx.y;
    const int b = bh / HH, h = bh % HH;
    const int qbase = blockIdx.x * 128 + wave * 16;
    const int qend  = qbase + 15;
    const size_t bS = (size_t)b * SS;
    const int pbase = wave * 512;
    const float scale = 0.125f;   // 1/sqrt(64)

    const _Float16* qrow = q16 + (bS + qbase + col) * EE + h * DD;
    v16h aq0 = ld_frag(qrow + ko,      qrow + ko + 16);
    v16h aq1 = ld_frag(qrow + 32 + ko, qrow + 32 + ko + 16);

    const _Float16* vtb = vt16 + ((size_t)b * HH + h) * DD * SS;

    v8f acc0 = {}, acc1 = {}, acc2 = {}, acc3 = {};
    float mrow[8], lrow[8];
#pragma unroll
    for (int r = 0; r < 8; ++r) { mrow[r] = -INFINITY; lrow[r] = 0.f; }

    for (int kk = 0; kk <= qend; kk += 32) {
        const _Float16* krow0 = k16 + (bS + kk + col) * EE + h * DD;
        const _Float16* krow1 = krow0 + (size_t)16 * EE;
        v8f s0 = {}, s1 = {};
        {
            v16h bk = ld_frag(krow0 + ko, krow0 + ko + 16);
            s0 = wmma_f16(aq0, bk, s0);
            bk = ld_frag(krow0 + 32 + ko, krow0 + 32 + ko + 16);
            s0 = wmma_f16(aq1, bk, s0);
            bk = ld_frag(krow1 + ko, krow1 + ko + 16);
            s1 = wmma_f16(aq0, bk, s1);
            bk = ld_frag(krow1 + 32 + ko, krow1 + 32 + ko + 16);
            s1 = wmma_f16(aq1, bk, s1);
        }
        const int key0 = kk + col, key1 = kk + 16 + col;

#pragma unroll
        for (int r = 0; r < 8; ++r) {
            const int qidx = qbase + half * 8 + r;
            float a0 = (key0 <= qidx) ? s0[r] * scale : -1e30f;
            float a1 = (key1 <= qidx) ? s1[r] * scale : -1e30f;
            float mx = fmaxf(a0, a1);
            mx = fmaxf(mx, __shfl_xor(mx, 1, 32));
            mx = fmaxf(mx, __shfl_xor(mx, 2, 32));
            mx = fmaxf(mx, __shfl_xor(mx, 4, 32));
            mx = fmaxf(mx, __shfl_xor(mx, 8, 32));
            const float nm = fmaxf(mrow[r], mx);
            const float al = __expf(mrow[r] - nm);
            const float p0 = __expf(a0 - nm);
            const float p1 = __expf(a1 - nm);
            float rs = p0 + p1;
            rs += __shfl_xor(rs, 1, 32);
            rs += __shfl_xor(rs, 2, 32);
            rs += __shfl_xor(rs, 4, 32);
            rs += __shfl_xor(rs, 8, 32);
            lrow[r] = lrow[r] * al + rs;
            mrow[r] = nm;
            acc0[r] *= al; acc1[r] *= al; acc2[r] *= al; acc3[r] *= al;
            pl[pbase + (half * 8 + r) * 32 + col]      = (_Float16)p0;
            pl[pbase + (half * 8 + r) * 32 + 16 + col] = (_Float16)p1;
        }

        // wave-private LDS tile: order writes before re-striped reads
        asm volatile("s_wait_dscnt 0" ::: "memory");

        v16h pa = ld_frag(&pl[pbase + col * 32 + ko], &pl[pbase + col * 32 + ko + 16]);

        {
            const _Float16* vr = vtb + (size_t)(col) * SS + kk;
            v16h vb = ld_frag(vr + ko, vr + ko + 16);
            acc0 = wmma_f16(pa, vb, acc0);
            vr = vtb + (size_t)(16 + col) * SS + kk;
            vb = ld_frag(vr + ko, vr + ko + 16);
            acc1 = wmma_f16(pa, vb, acc1);
            vr = vtb + (size_t)(32 + col) * SS + kk;
            vb = ld_frag(vr + ko, vr + ko + 16);
            acc2 = wmma_f16(pa, vb, acc2);
            vr = vtb + (size_t)(48 + col) * SS + kk;
            vb = ld_frag(vr + ko, vr + ko + 16);
            acc3 = wmma_f16(pa, vb, acc3);
        }
    }

#pragma unroll
    for (int r = 0; r < 8; ++r) {
        const int q = qbase + half * 8 + r;
        const float inv = 1.0f / lrow[r];
        _Float16* orow = attn16 + (bS + q) * EE + h * DD;
        orow[col]      = (_Float16)(acc0[r] * inv);
        orow[16 + col] = (_Float16)(acc1[r] * inv);
        orow[32 + col] = (_Float16)(acc2[r] * inv);
        orow[48 + col] = (_Float16)(acc3[r] * inv);
    }
}

// ---------------- host launcher ----------------
extern "C" void kernel_launch(void* const* d_in, const int* in_sizes, int n_in,
                              void* d_out, int out_size, void* d_ws, size_t ws_size,
                              hipStream_t stream) {
    (void)in_sizes; (void)n_in; (void)out_size; (void)ws_size;
    const float* x     = (const float*)d_in[0];
    const float* wq    = (const float*)d_in[1];
    const float* wk    = (const float*)d_in[2];
    const float* wv    = (const float*)d_in[3];
    const float* wo    = (const float*)d_in[4];
    const float* bo    = (const float*)d_in[5];
    const float* w1    = (const float*)d_in[6];
    const float* b1    = (const float*)d_in[7];
    const float* w2    = (const float*)d_in[8];
    const float* b2    = (const float*)d_in[9];
    const float* gamma = (const float*)d_in[10];
    const float* beta  = (const float*)d_in[11];
    float* out = (float*)d_out;

    const size_t nEE = (size_t)EE * EE;
    const size_t nEF = (size_t)EE * FFF;
    const size_t nME = (size_t)MM * EE;
    const size_t nMF = (size_t)MM * FFF;

    char* ws = (char*)d_ws;
    size_t off = 0;
    auto alloc = [&](size_t bytes) -> void* {
        void* p = ws + off;
        off = (off + bytes + 255) & ~(size_t)255;
        return p;
    };
    _Float16* wq16   = (_Float16*)alloc(nEE * 2);
    _Float16* wk16   = (_Float16*)alloc(nEE * 2);
    _Float16* wv16   = (_Float16*)alloc(nEE * 2);
    _Float16* wo16   = (_Float16*)alloc(nEE * 2);
    _Float16* w116   = (_Float16*)alloc(nEF * 2);
    _Float16* w216   = (_Float16*)alloc(nEF * 2);
    _Float16* xn16   = (_Float16*)alloc(nME * 2);
    _Float16* q16    = (_Float16*)alloc(nME * 2);
    _Float16* k16    = (_Float16*)alloc(nME * 2);
    _Float16* vt16   = (_Float16*)alloc(nME * 2);
    _Float16* attn16 = (_Float16*)alloc(nME * 2);
    _Float16* y16    = (_Float16*)alloc(nME * 2);
    _Float16* h16    = (_Float16*)alloc(nMF * 2);
    float*    res    = (float*)alloc(nME * 4);

    cvt_f16_kernel<<<(nEE + 255) / 256, 256, 0, stream>>>(wq, wq16, (int)nEE);
    cvt_f16_kernel<<<(nEE + 255) / 256, 256, 0, stream>>>(wk, wk16, (int)nEE);
    cvt_f16_kernel<<<(nEE + 255) / 256, 256, 0, stream>>>(wv, wv16, (int)nEE);
    cvt_f16_kernel<<<(nEE + 255) / 256, 256, 0, stream>>>(wo, wo16, (int)nEE);
    cvt_f16_kernel<<<(nEF + 255) / 256, 256, 0, stream>>>(w1, w116, (int)nEF);
    cvt_f16_kernel<<<(nEF + 255) / 256, 256, 0, stream>>>(w2, w216, (int)nEF);

    ln_f16_kernel<<<MM, 256, 0, stream>>>(x, gamma, beta, xn16);

    dim3 gEE(EE / 64, MM / 128);
    gemm_wmma_kernel<<<gEE, 256, 0, stream>>>(xn16, wq16, nullptr, nullptr, nullptr, q16,  MM, EE, EE, 0);
    gemm_wmma_kernel<<<gEE, 256, 0, stream>>>(xn16, wk16, nullptr, nullptr, nullptr, k16,  MM, EE, EE, 0);
    gemm_wmma_kernel<<<gEE, 256, 0, stream>>>(xn16, wv16, nullptr, nullptr, nullptr, vt16, MM, EE, EE, 1);

    flash_attn_kernel<<<dim3(SS / 128, BB * HH), 256, 0, stream>>>(q16, k16, vt16, attn16);

    gemm_wmma_kernel<<<gEE, 256, 0, stream>>>(attn16, wo16, bo, x, res, nullptr, MM, EE, EE, 2);

    ln_f16_kernel<<<MM, 256, 0, stream>>>(res, gamma, beta, y16);

    dim3 gFF(FFF / 64, MM / 128);
    gemm_wmma_kernel<<<gFF, 256, 0, stream>>>(y16, w116, b1, nullptr, nullptr, h16, MM, FFF, EE, 3);
    gemm_wmma_kernel<<<gEE, 256, 0, stream>>>(h16, w216, b2, res, out, nullptr, MM, EE, FFF, 2);
}
